// SConv2d_v3_27144193311463
// MI455X (gfx1250) — matrix operand order repaired and checked by reading, not compiled, not run on hardware
//
#include <hip/hip_runtime.h>

typedef __attribute__((ext_vector_type(16))) _Float16 v16h;
typedef __attribute__((ext_vector_type(8)))  _Float16 v8h;
typedef __attribute__((ext_vector_type(8)))  float    v8f;

#define SHUF16(a,b) __builtin_shufflevector(a,b,0,1,2,3,4,5,6,7,8,9,10,11,12,13,14,15)

#define Bn 8
#define CIN 64
#define COUT 64
#define Hn 128
#define Wn 128
#define HW (Hn*Wn)
#define PSTR 72   // padded LDS row stride in halves (144B: 16B aligned, bank-spread)

// ---------------- guide path ----------------

__global__ __launch_bounds__(256) void k_sig(const float* __restrict__ pred,
                                             float* __restrict__ sig) {
    int i = blockIdx.x * 256 + threadIdx.x;
    if (i < Bn * HW) sig[i] = 1.f / (1.f + expf(-pred[i]));
}

__global__ __launch_bounds__(256) void k_blur(const float* __restrict__ sig,
                                              const float* __restrict__ gk,
                                              float* __restrict__ bl) {
    int i = blockIdx.x * 256 + threadIdx.x;
    if (i >= Bn * HW) return;
    int b = i >> 14, p = i & (HW - 1);
    int y = p >> 7, x = p & 127;
    const float* s = sig + (b << 14);
    float acc = 0.f;
    for (int dy = 0; dy < 15; ++dy) {
        int yy = y + dy - 7;
        if (yy < 0 || yy > 127) continue;
        for (int dx = 0; dx < 15; ++dx) {
            int xx = x + dx - 7;
            if (xx < 0 || xx > 127) continue;
            acc += gk[dy * 15 + dx] * s[(yy << 7) + xx];
        }
    }
    bl[i] = acc;
}

__global__ __launch_bounds__(256) void k_minmax(const float* __restrict__ bl,
                                                float* __restrict__ mnmx) {
    __shared__ float smn[256], smx[256];
    int b = blockIdx.x, t = threadIdx.x;
    const float* p = bl + (b << 14);
    float mn = 3.4e38f, mx = -3.4e38f;
    for (int i = t; i < HW; i += 256) {
        float v = p[i];
        mn = fminf(mn, v); mx = fmaxf(mx, v);
    }
    smn[t] = mn; smx[t] = mx;
    __syncthreads();
    for (int s = 128; s > 0; s >>= 1) {
        if (t < s) { smn[t] = fminf(smn[t], smn[t + s]); smx[t] = fmaxf(smx[t], smx[t + s]); }
        __syncthreads();
    }
    if (t == 0) { mnmx[b * 2] = smn[0]; mnmx[b * 2 + 1] = smx[0]; }
}

__global__ __launch_bounds__(256) void k_mask(const float* __restrict__ bl,
                                              const float* __restrict__ sig,
                                              const float* __restrict__ mnmx,
                                              unsigned char* __restrict__ mask) {
    int i = blockIdx.x * 256 + threadIdx.x;
    if (i >= Bn * HW) return;
    int b = i >> 14;
    float mn = mnmx[b * 2], mx = mnmx[b * 2 + 1];
    float sa = (bl[i] - mn) / (mx - mn + 1e-8f);
    float h  = fmaxf(sa, sig[i]);
    mask[i] = (h >= 0.5f) ? 0 : 1;   // argmax([h, 1-h]) with tie -> 0
}

// ---------------- dynamic weight generation ----------------

// adaptive_avg_pool with PyTorch bins: [floor(i*n/k), ceil((i+1)*n/k))
__global__ __launch_bounds__(64) void k_pool(const float* __restrict__ feat,
                                             float* __restrict__ pooled) {
    __shared__ float red[64];
    int blk = blockIdx.x;               // (b*64+ci)*9 + bin
    int bin = blk % 9, bc = blk / 9;
    int iy = bin / 3, ix = bin % 3;
    int rs = (iy * Hn) / 3, re = ((iy + 1) * Hn + 2) / 3;
    int cs = (ix * Wn) / 3, ce = ((ix + 1) * Wn + 2) / 3;
    int nr = re - rs, nc = ce - cs;
    const float* p = feat + (size_t)bc * HW;
    float s = 0.f;
    for (int idx = threadIdx.x; idx < nr * nc; idx += 64) {
        int ry = idx / nc, rx = idx - ry * nc;
        s += p[(rs + ry) * Wn + cs + rx];
    }
    red[threadIdx.x] = s;
    __syncthreads();
    for (int t = 32; t > 0; t >>= 1) {
        if (threadIdx.x < t) red[threadIdx.x] += red[threadIdx.x + t];
        __syncthreads();
    }
    if (threadIdx.x == 0) pooled[blk] = red[0] / (float)(nr * nc);
}

__global__ __launch_bounds__(256) void k_mlp1(const float* __restrict__ pooled,
                                              const float* __restrict__ w1,
                                              const float* __restrict__ b1,
                                              float* __restrict__ h1) {
    int i = blockIdx.x * 256 + threadIdx.x;    // (b*16+o)*9 + pos
    if (i >= Bn * 16 * 9) return;
    int pos = i % 9, t = i / 9;
    int o = t % 16, b = t / 16;
    float s = b1[o];
    for (int ci = 0; ci < CIN; ++ci)
        s += w1[o * CIN + ci] * pooled[(b * CIN + ci) * 9 + pos];
    h1[i] = 1.f / (1.f + expf(-s));
}

// Produce A-fragments pre-swizzled for v_wmma_f32_16x16x32_f16 (hi/lo f16 split).
// K ordering: k = (dy*3+dx)*64 + ci. One thread per (b,r,mtile,kchunk,lane).
__global__ __launch_bounds__(256) void k_pack(const float* __restrict__ h1,
                                              const float* __restrict__ w2,
                                              const float* __restrict__ b2,
                                              v8h* __restrict__ Ahi,
                                              v8h* __restrict__ Alo) {
    int tid = blockIdx.x * 256 + threadIdx.x;
    if (tid >= Bn * 2 * 4 * 18 * 32) return;
    int lane = tid & 31; int t = tid >> 5;
    int kc = t % 18; t /= 18;
    int mtile = t & 3; t >>= 2;
    int r = t & 1; int b = t >> 1;

    int off    = kc >> 1;          // tap index dy*3+dx (K-chunk of 32 stays in one 64-block)
    int cibase = (kc & 1) * 32;
    int co     = mtile * 16 + (lane & 15);
    bool hiL   = lane >= 16;

    float hv[8];
#pragma unroll
    for (int i = 0; i < 8; ++i) hv[i] = h1[(b * 16 + r * 8 + i) * 9 + off];

    _Float16 hh[16], ll[16];
#pragma unroll
    for (int h = 0; h < 16; ++h) {
        // A 16x32 layout (contiguous K, matches B): lanes<16: K = h ; lanes>=16: K = h+16
        int Kl = (hiL ? 16 : 0) + h;
        int ci = cibase + Kl;
        int widx = r * (CIN * COUT) + co * CIN + ci;
        float w = b2[widx];
        const float* wp = w2 + (size_t)widx * 8;
#pragma unroll
        for (int i = 0; i < 8; ++i) w += wp[i] * hv[i];
        _Float16 fh = (_Float16)w;
        hh[h] = fh;
        ll[h] = (_Float16)(w - (float)fh);
    }
    int aIdx = (((((b * 2 + r) * 4 + mtile) * 18) + kc) * 32 + lane) * 2;
    v8h v0 = {}, v1 = {}, u0 = {}, u1 = {};
#pragma unroll
    for (int j = 0; j < 8; ++j) { v0[j] = hh[j]; v1[j] = hh[j + 8]; u0[j] = ll[j]; u1[j] = ll[j + 8]; }
    Ahi[aIdx] = v0; Ahi[aIdx + 1] = v1;
    Alo[aIdx] = u0; Alo[aIdx + 1] = u1;
}

// ---------------- main implicit-GEMM conv + branch selection ----------------

__global__ __launch_bounds__(256) void k_main(const float* __restrict__ feat,
                                              const v8h* __restrict__ Ahi,
                                              const v8h* __restrict__ Alo,
                                              const unsigned char* __restrict__ mask,
                                              float* __restrict__ out) {
    __shared__ _Float16 phi[100 * PSTR];
    __shared__ _Float16 plo[100 * PSTR];

    int blk = blockIdx.x;
    int b = blk >> 8, t = blk & 255;
    int ty = t >> 4, tx = t & 15;
    int y0 = ty * 8, x0 = tx * 8;
    int tid = threadIdx.x;

    // Stage 10x10x64 halo patch into LDS as f16 hi/lo (zero pad at image edges)
    const float* fb = feat + (size_t)b * CIN * HW;
    for (int idx = tid; idx < 6400; idx += 256) {
        int ci = idx / 100, pos = idx - ci * 100;
        int py = pos / 10, px = pos - py * 10;
        int gy = y0 + py - 1, gx = x0 + px - 1;
        float v = 0.f;
        if (gy >= 0 && gy < Hn && gx >= 0 && gx < Wn)
            v = fb[(ci * Hn + gy) * Wn + gx];
        _Float16 h = (_Float16)v;
        phi[pos * PSTR + ci] = h;
        plo[pos * PSTR + ci] = (_Float16)(v - (float)h);
    }
    __syncthreads();

    int lane   = tid & 31;
    int wave   = tid >> 5;
    int r      = wave >> 2;      // branch
    int mtile  = wave & 3;       // 16 couts per tile
    int ln15   = lane & 15;
    int hiHalf = (lane >> 4) & 1;

    v8f acc0 = {}, acc1 = {}, acc2 = {}, acc3 = {};

    for (int kc = 0; kc < 18; ++kc) {
        int off = kc >> 1;
        int dy = off / 3, dx = off - dy * 3;
        int ci0 = (kc & 1) * 32 + hiHalf * 16;

        int aIdx = (((((b * 2 + r) * 4 + mtile) * 18) + kc) * 32 + lane) * 2;
        v8h ah0 = Ahi[aIdx], ah1 = Ahi[aIdx + 1];
        v8h al0 = Alo[aIdx], al1 = Alo[aIdx + 1];
        v16h ah = SHUF16(ah0, ah1);
        v16h al = SHUF16(al0, al1);

#pragma unroll
        for (int nt = 0; nt < 4; ++nt) {
            int p  = nt * 16 + ln15;                 // pixel column N = lane&15
            int py = p >> 3, px = p & 7;
            int pos = (py + dy) * 10 + (px + dx);
            const _Float16* bph = &phi[pos * PSTR + ci0];
            const _Float16* bpl = &plo[pos * PSTR + ci0];
            v8h bh0 = *(const v8h*)bph, bh1 = *(const v8h*)(bph + 8);
            v8h bl0 = *(const v8h*)bpl, bl1 = *(const v8h*)(bpl + 8);
            v16h bh = SHUF16(bh0, bh1);
            v16h bl = SHUF16(bl0, bl1);
            v8f* accp = (nt == 0) ? &acc0 : (nt == 1) ? &acc1 : (nt == 2) ? &acc2 : &acc3;
            v8f c = *accp;
            c = __builtin_amdgcn_wmma_f32_16x16x32_f16(false, ah, false, bh, (short)0, c, false, false);
            c = __builtin_amdgcn_wmma_f32_16x16x32_f16(false, ah, false, bl, (short)0, c, false, false);
            c = __builtin_amdgcn_wmma_f32_16x16x32_f16(false, al, false, bh, (short)0, c, false, false);
            *accp = c;
        }
    }

    // Branch-select store: the wave whose r matches the per-pixel mask owns the write.
    const unsigned char* mb = mask + b * HW;
    float* ob = out + (size_t)b * COUT * HW;
    int coBase = mtile * 16 + hiHalf * 8;   // C/D layout: lanes<16 -> M=v, lanes>=16 -> M=8+v
#pragma unroll
    for (int nt = 0; nt < 4; ++nt) {
        int p = nt * 16 + ln15;
        int y = y0 + (p >> 3), x = x0 + (p & 7);
        if ((int)mb[y * Wn + x] == r) {
            v8f c = (nt == 0) ? acc0 : (nt == 1) ? acc1 : (nt == 2) ? acc2 : acc3;
#pragma unroll
            for (int v = 0; v < 8; ++v)
                ob[((coBase + v) * Hn + y) * Wn + x] = c[v];
        }
    }
}

// ---------------- launch ----------------

extern "C" void kernel_launch(void* const* d_in, const int* in_sizes, int n_in,
                              void* d_out, int out_size, void* d_ws, size_t ws_size,
                              hipStream_t stream) {
    (void)in_sizes; (void)n_in; (void)out_size; (void)ws_size;
    const float* feat = (const float*)d_in[0];
    const float* pred = (const float*)d_in[1];
    const float* w1   = (const float*)d_in[2];
    const float* b1   = (const float*)d_in[3];
    const float* w2   = (const float*)d_in[4];
    const float* b2   = (const float*)d_in[5];
    const float* gk   = (const float*)d_in[6];
    float* out = (float*)d_out;

    char* ws = (char*)d_ws;
    size_t o = 0;
    auto carve = [&](size_t bytes) -> char* {
        char* p = ws + o;
        o = (o + bytes + 255) & ~(size_t)255;
        return p;
    };
    float* sig    = (float*)carve((size_t)Bn * HW * 4);
    float* bl     = (float*)carve((size_t)Bn * HW * 4);
    float* mnmx   = (float*)carve(64);
    float* pooled = (float*)carve((size_t)Bn * CIN * 9 * 4);
    float* h1     = (float*)carve((size_t)Bn * 16 * 9 * 4);
    unsigned char* mask = (unsigned char*)carve((size_t)Bn * HW);
    v8h* Ahi = (v8h*)carve((size_t)Bn * 2 * 4 * 18 * 32 * 32);
    v8h* Alo = (v8h*)carve((size_t)Bn * 2 * 4 * 18 * 32 * 32);

    // guide path
    k_sig   <<<512, 256, 0, stream>>>(pred, sig);
    k_blur  <<<512, 256, 0, stream>>>(sig, gk, bl);
    k_minmax<<<Bn, 256, 0, stream>>>(bl, mnmx);
    k_mask  <<<512, 256, 0, stream>>>(bl, sig, mnmx, mask);

    // dynamic weights
    k_pool<<<Bn * CIN * 9, 64, 0, stream>>>(feat, pooled);
    k_mlp1<<<(Bn * 16 * 9 + 255) / 256, 256, 0, stream>>>(pooled, w1, b1, h1);
    k_pack<<<(Bn * 2 * 4 * 18 * 32) / 256, 256, 0, stream>>>(h1, w2, b2, Ahi, Alo);

    // main WMMA conv + branch select
    k_main<<<Bn * (Hn / 8) * (Wn / 8), 256, 0, stream>>>(feat, Ahi, Alo, mask, out);
}
